// TDDMN_50259707297882
// MI455X (gfx1250) — matrix operand order repaired and checked by reading, not compiled, not run on hardware
//
#include <hip/hip_runtime.h>
#include <hip/hip_bf16.h>

// ---------------------------------------------------------------------------
// TDDMN (DMN episodic memory) for gfx1250: bf16 WMMA everywhere.
//   N=16, S=64, H=256, FA=256
// ---------------------------------------------------------------------------

typedef __attribute__((ext_vector_type(16))) __bf16 v16bf;
typedef __attribute__((ext_vector_type(8)))  __bf16 v8bf;
typedef __attribute__((ext_vector_type(8)))  float  v8f;

__device__ __forceinline__ unsigned short f2bf(float f) {
    unsigned int u = __float_as_uint(f);
    u += 0x7FFFu + ((u >> 16) & 1u);      // round-to-nearest-even
    return (unsigned short)(u >> 16);
}

union V16U { v16bf v; v8bf h[2]; };

// A-fragment (16 rows x 32 K, bf16) from an LDS tile with row stride 40 elems.
// ISA layout: lanes 0-15 -> K=0..7 & 16..23, lanes 16-31 -> K=8..15 & 24..31.
__device__ __forceinline__ v16bf load_a_frag(const unsigned short* z, int rowBase, int lane) {
    int row = rowBase + (lane & 15);
    int ko  = (lane >> 4) << 3;
    const unsigned short* p = z + row * 40 + ko;
    V16U u;
    u.h[0] = *(const v8bf*)(p);        // K = ko .. ko+7
    u.h[1] = *(const v8bf*)(p + 16);   // K = ko+16 .. ko+23
    return u.v;
}

// B-fragment: pre-swizzled in workspace, 512 bf16 per (32K x 16N) tile,
// lane-contiguous (lane*16 + e). One 32-byte load per lane.
__device__ __forceinline__ v16bf load_b_frag(const unsigned short* wsB, int tile, int lane) {
    return *(const v16bf*)(wsB + (size_t)tile * 512 + lane * 16);
}

__device__ __forceinline__ v8f wmma_bf16(v16bf a, v16bf b, v8f c) {
    return __builtin_amdgcn_wmma_f32_16x16x32_bf16(false, a, false, b, (short)0, c, false, false);
}

// ---------------------------------------------------------------------------
// Prep: convert f32 weights to bf16 in B-fragment order.
// dst[tile*512 + lane*16 + e], tile = nb*(K/32)+kb, B[k][n] = src[n][k].
// Two sources concatenated along N (src1 used for n >= nSplit).
// ---------------------------------------------------------------------------
__global__ void __launch_bounds__(256) swz_kernel(
    const float* __restrict__ src0, const float* __restrict__ src1,
    int nSplit, int K, int Ntot, unsigned short* __restrict__ dst) {
    int idx = blockIdx.x * 256 + threadIdx.x;
    if (idx >= K * Ntot) return;
    int e    = idx & 15;
    int lane = (idx >> 4) & 31;
    int tile = idx >> 9;
    int KB = K >> 5;
    int kb = tile % KB;
    int nb = tile / KB;
    int nIdx = nb * 16 + (lane & 15);
    int kIdx = kb * 32 + ((lane >> 4) << 3) + (e & 7) + ((e >> 3) << 4);
    const float* s = src0;
    if (nIdx >= nSplit) { s = src1; nIdx -= nSplit; }
    dst[idx] = f2bf(s[nIdx * K + kIdx]);
}

// ---------------------------------------------------------------------------
// Gate + softmax. One block per (n,i): rows j=0..63 of the (64 x 1024) z
// matrix, GEMM vs z1_w^T (1024x256), tanh, dot with z2, masked softmax.
// ---------------------------------------------------------------------------
__global__ void __launch_bounds__(256) gate_kernel(
    const float* __restrict__ facts, const float* __restrict__ prevM,
    const float* __restrict__ questions, const int* __restrict__ doc_len,
    const unsigned short* __restrict__ z1_swz, const float* __restrict__ z1_b,
    const float* __restrict__ z2_w, const float* __restrict__ z2_b,
    float* __restrict__ attn_out) {
    __shared__ __align__(16) unsigned short ztile[64 * 40];
    __shared__ float qv[256], mv[256], z2v[256], z1bv[256], Gl[64];

    int tid = threadIdx.x, lane = tid & 31, w = tid >> 5;
    int n = blockIdx.x >> 6, i = blockIdx.x & 63;

    qv[tid]   = questions[((n * 64 + i) << 8) + tid];
    mv[tid]   = prevM[((n * 64 + i) << 8) + tid];
    z2v[tid]  = z2_w[tid];
    z1bv[tid] = z1_b[tid];
    if (tid < 64) Gl[tid] = 0.f;
    __syncthreads();

    int stripe = w >> 1;           // 16-row stripe of j
    int cbase  = (w & 1) * 8;      // 8 column tiles of 16
    v8f acc[8];
#pragma unroll
    for (int t = 0; t < 8; ++t)
#pragma unroll
        for (int r = 0; r < 8; ++r) acc[t][r] = 0.f;

    const float* fbase = facts + ((size_t)n << 14);
    for (int kb = 0; kb < 32; ++kb) {
        int p  = kb >> 3;
        int hb = (kb & 7) << 5;
        // build 64x32 bf16 z tile for this K block
#pragma unroll
        for (int l = 0; l < 8; ++l) {
            int e = (tid << 3) + l;
            int j = e >> 5, kk = e & 31, h = hb + kk;
            float fv = fbase[(j << 8) + h];
            float zv;
            if (p == 0)      zv = fv * qv[h];
            else if (p == 1) zv = fv * mv[h];
            else if (p == 2) zv = fabsf(fv - qv[h]);
            else             zv = fabsf(fv - mv[h]);
            ztile[j * 40 + kk] = f2bf(zv);
        }
        __syncthreads();
        v16bf a = load_a_frag(ztile, stripe * 16, lane);
#pragma unroll
        for (int t = 0; t < 8; ++t) {
            int nb = cbase + t;
            if (kb < 31)
                __builtin_prefetch(z1_swz + (size_t)(nb * 32 + kb + 1) * 512 + lane * 16, 0, 1);
            v16bf b = load_b_frag(z1_swz, nb * 32 + kb, lane);
            acc[t] = wmma_bf16(a, b, acc[t]);
        }
        __syncthreads();
    }

    // Gh = tanh(acc + z1_b); G = Gh @ z2^T   (reduce 256 cols per row)
    float part[8];
#pragma unroll
    for (int r = 0; r < 8; ++r) part[r] = 0.f;
#pragma unroll
    for (int t = 0; t < 8; ++t) {
        int col = (cbase + t) * 16 + (lane & 15);
#pragma unroll
        for (int r = 0; r < 8; ++r)
            part[r] += tanhf(acc[t][r] + z1bv[col]) * z2v[col];
    }
#pragma unroll
    for (int r = 0; r < 8; ++r) {
        int rowl = stripe * 16 + r + ((lane >> 4) << 3);
        atomicAdd(&Gl[rowl], part[r]);      // ds_add_f32
    }
    __syncthreads();

    if (tid == 0) {
        int dlen = doc_len[n];
        float z2bs = z2_b[0];
        float mx = -3.4e38f;
        for (int j = 0; j < dlen; ++j) mx = fmaxf(mx, Gl[j] + z2bs);
        float sum = 0.f;
        for (int j = 0; j < dlen; ++j) sum += __expf(Gl[j] + z2bs - mx);
        float inv = 1.f / sum;
        float* orow = attn_out + ((n * 64 + i) << 6);
        for (int j = 0; j < 64; ++j)
            orow[j] = (j < dlen) ? __expf(Gl[j] + z2bs - mx) * inv : 0.f;
    }
}

// ---------------------------------------------------------------------------
// Fact projections: Aproj[n*64+t][0:256]   = facts@Wr^T + Wr_b + Ur_b
//                   Aproj[n*64+t][256:512] = facts@W^T  + W_b
// (1024 x 256) @ (256 x 512); block = (n, half of 32 t-rows).
// ---------------------------------------------------------------------------
__global__ void __launch_bounds__(256) factproj_kernel(
    const float* __restrict__ facts, const unsigned short* __restrict__ wrw_swz,
    const float* __restrict__ Wr_b, const float* __restrict__ Ur_b,
    const float* __restrict__ W_b, float* __restrict__ Aproj) {
    __shared__ __align__(16) unsigned short ztile[32 * 40];
    int tid = threadIdx.x, lane = tid & 31, w = tid >> 5;
    int n = blockIdx.x >> 1, half = blockIdx.x & 1;
    int stripe = w >> 2, base = (w & 3) * 8;

    v8f acc[8];
#pragma unroll
    for (int t = 0; t < 8; ++t)
#pragma unroll
        for (int r = 0; r < 8; ++r) acc[t][r] = 0.f;

    const float* fbase = facts + ((size_t)(n * 64 + half * 32) << 8);
    for (int kb = 0; kb < 8; ++kb) {
#pragma unroll
        for (int l = 0; l < 4; ++l) {
            int e = (tid << 2) + l;
            int j = e >> 5, kk = e & 31;
            ztile[j * 40 + kk] = f2bf(fbase[(j << 8) + (kb << 5) + kk]);
        }
        __syncthreads();
        v16bf a = load_a_frag(ztile, stripe * 16, lane);
#pragma unroll
        for (int t = 0; t < 8; ++t) {
            v16bf b = load_b_frag(wrw_swz, (base + t) * 8 + kb, lane);
            acc[t] = wmma_bf16(a, b, acc[t]);
        }
        __syncthreads();
    }
#pragma unroll
    for (int t = 0; t < 8; ++t) {
        int col = (base + t) * 16 + (lane & 15);
        float bias = (col < 256) ? (Wr_b[col] + Ur_b[col]) : W_b[col - 256];
#pragma unroll
        for (int r = 0; r < 8; ++r) {
            int rowl = stripe * 16 + r + ((lane >> 4) << 3);
            int grow = n * 64 + half * 32 + rowl;
            Aproj[(size_t)grow * 512 + col] = acc[t][r] + bias;
        }
    }
}

// ---------------------------------------------------------------------------
// Sequential GRU-style scan. Rows are independent -> block owns 16 batch
// rows (n, ig*16..+15), keeps C in LDS, loops t=0..63. Per step:
// (16x256)@(256x512) via WMMA (Ur and U paired in the same wave so the
// r/h/C update is pure register math).
// ---------------------------------------------------------------------------
__global__ void __launch_bounds__(256) recur_kernel(
    const float* __restrict__ Aproj, const unsigned short* __restrict__ uru_swz,
    const float* __restrict__ U_b, const float* __restrict__ attn,
    float* __restrict__ C_ws) {
    __shared__ float Cm[16][260];
    __shared__ __align__(16) unsigned short czb[16 * 40];
    __shared__ float Av[256], Awv[256], Ubv[256], gvec[16];

    int tid = threadIdx.x, lane = tid & 31, w = tid >> 5;
    int n = blockIdx.x >> 2, ig = blockIdx.x & 3;

    Ubv[tid] = U_b[tid];
#pragma unroll
    for (int l = 0; l < 16; ++l) {
        int e = tid * 16 + l;
        Cm[e >> 8][e & 255] = 0.f;
    }
    __syncthreads();

    for (int t = 0; t < 64; ++t) {
        Av[tid]  = Aproj[(size_t)(n * 64 + t) * 512 + tid];
        Awv[tid] = Aproj[(size_t)(n * 64 + t) * 512 + 256 + tid];
        if (tid < 16) gvec[tid] = attn[((n * 64 + ig * 16 + tid) << 6) + t];
        __syncthreads();

        v8f accR[2], accU[2];
#pragma unroll
        for (int jj = 0; jj < 2; ++jj)
#pragma unroll
            for (int r = 0; r < 8; ++r) { accR[jj][r] = 0.f; accU[jj][r] = 0.f; }

        for (int kb = 0; kb < 8; ++kb) {
#pragma unroll
            for (int l = 0; l < 2; ++l) {
                int e = (tid << 1) + l;
                int j = e >> 5, kk = e & 31;
                czb[j * 40 + kk] = f2bf(Cm[j][(kb << 5) + kk]);
            }
            __syncthreads();
            v16bf a = load_a_frag(czb, 0, lane);
#pragma unroll
            for (int jj = 0; jj < 2; ++jj) {
                int ncR = w * 2 + jj;           // Ur columns
                int ncU = 16 + ncR;             // U  columns
                v16bf bR = load_b_frag(uru_swz, ncR * 8 + kb, lane);
                v16bf bU = load_b_frag(uru_swz, ncU * 8 + kb, lane);
                accR[jj] = wmma_bf16(a, bR, accR[jj]);
                accU[jj] = wmma_bf16(a, bU, accU[jj]);
            }
            __syncthreads();
        }

#pragma unroll
        for (int jj = 0; jj < 2; ++jj) {
            int col = (w * 2 + jj) * 16 + (lane & 15);
#pragma unroll
            for (int r = 0; r < 8; ++r) {
                int rowl = r + ((lane >> 4) << 3);
                float rr = 1.f / (1.f + __expf(-(Av[col] + accR[jj][r])));
                float hh = tanhf(Awv[col] + rr * (accU[jj][r] + Ubv[col]));
                float g  = gvec[rowl];
                Cm[rowl][col] = g * hh + (1.f - g) * Cm[rowl][col];
            }
        }
        __syncthreads();
    }

#pragma unroll
    for (int l = 0; l < 16; ++l) {
        int e = tid * 16 + l;
        int j = e >> 8, col = e & 255;
        C_ws[(size_t)(n * 64 + ig * 16 + j) * 256 + col] = Cm[j][col];
    }
}

// ---------------------------------------------------------------------------
// next_mem = relu([prevM, C, questions] @ nm_w^T + nm_b): (1024x768)@(768x256)
// ---------------------------------------------------------------------------
__global__ void __launch_bounds__(256) nextmem_kernel(
    const float* __restrict__ prevM, const float* __restrict__ C_ws,
    const float* __restrict__ questions, const unsigned short* __restrict__ nm_swz,
    const float* __restrict__ nm_b, float* __restrict__ out) {
    __shared__ __align__(16) unsigned short ztile[64 * 40];
    int tid = threadIdx.x, lane = tid & 31, w = tid >> 5;
    int n = blockIdx.x;
    int stripe = w >> 1, cbase = (w & 1) * 8;

    v8f acc[8];
#pragma unroll
    for (int t = 0; t < 8; ++t)
#pragma unroll
        for (int r = 0; r < 8; ++r) acc[t][r] = 0.f;

    for (int kb = 0; kb < 24; ++kb) {
        int seg = kb >> 3;
        int hb  = (kb & 7) << 5;
        const float* src = (seg == 0) ? prevM : (seg == 1 ? C_ws : questions);
#pragma unroll
        for (int l = 0; l < 8; ++l) {
            int e = (tid << 3) + l;
            int j = e >> 5, kk = e & 31;
            ztile[j * 40 + kk] = f2bf(src[(size_t)(n * 64 + j) * 256 + hb + kk]);
        }
        __syncthreads();
        v16bf a = load_a_frag(ztile, stripe * 16, lane);
#pragma unroll
        for (int t = 0; t < 8; ++t) {
            v16bf b = load_b_frag(nm_swz, (cbase + t) * 24 + kb, lane);
            acc[t] = wmma_bf16(a, b, acc[t]);
        }
        __syncthreads();
    }
#pragma unroll
    for (int t = 0; t < 8; ++t) {
        int col = (cbase + t) * 16 + (lane & 15);
        float bias = nm_b[col];
#pragma unroll
        for (int r = 0; r < 8; ++r) {
            int row = stripe * 16 + r + ((lane >> 4) << 3);
            out[(size_t)(n * 64 + row) * 256 + col] = fmaxf(acc[t][r] + bias, 0.f);
        }
    }
}

// ---------------------------------------------------------------------------
extern "C" void kernel_launch(void* const* d_in, const int* in_sizes, int n_in,
                              void* d_out, int out_size, void* d_ws, size_t ws_size,
                              hipStream_t stream) {
    const float* facts     = (const float*)d_in[0];
    const float* prevM     = (const float*)d_in[1];
    const float* questions = (const float*)d_in[2];
    const int*   doc_len   = (const int*)d_in[3];
    const float* z1_w = (const float*)d_in[4];
    const float* z1_b = (const float*)d_in[5];
    const float* z2_w = (const float*)d_in[6];
    const float* z2_b = (const float*)d_in[7];
    const float* Wr_w = (const float*)d_in[8];
    const float* Wr_b = (const float*)d_in[9];
    const float* Ur_w = (const float*)d_in[10];
    const float* Ur_b = (const float*)d_in[11];
    const float* W_w  = (const float*)d_in[12];
    const float* W_b  = (const float*)d_in[13];
    const float* U_w  = (const float*)d_in[14];
    const float* U_b  = (const float*)d_in[15];
    const float* nm_w = (const float*)d_in[16];
    const float* nm_b = (const float*)d_in[17];

    char* ws = (char*)d_ws;
    unsigned short* z1_swz  = (unsigned short*)(ws);                 // 512 KB
    unsigned short* wrw_swz = (unsigned short*)(ws + 524288);        // 256 KB
    unsigned short* uru_swz = (unsigned short*)(ws + 786432);        // 256 KB
    unsigned short* nm_swz  = (unsigned short*)(ws + 1048576);       // 384 KB
    float* Aproj = (float*)(ws + 1441792);                           // 2 MB
    float* C_ws  = (float*)(ws + 1441792 + 2097152);                 // 1 MB

    float* next_mem = (float*)d_out;                 // (16,64,256)
    float* attn     = next_mem + 16 * 64 * 256;      // (16,64,64)

    // weight swizzle (bf16, B-fragment layout)
    swz_kernel<<<(1024 * 256) / 256, 256, 0, stream>>>(z1_w, z1_w, 256, 1024, 256, z1_swz);
    swz_kernel<<<(256 * 512) / 256, 256, 0, stream>>>(Wr_w, W_w, 256, 256, 512, wrw_swz);
    swz_kernel<<<(256 * 512) / 256, 256, 0, stream>>>(Ur_w, U_w, 256, 256, 512, uru_swz);
    swz_kernel<<<(768 * 256) / 256, 256, 0, stream>>>(nm_w, nm_w, 256, 768, 256, nm_swz);

    gate_kernel<<<1024, 256, 0, stream>>>(facts, prevM, questions, doc_len,
                                          z1_swz, z1_b, z2_w, z2_b, attn);
    factproj_kernel<<<32, 256, 0, stream>>>(facts, wrw_swz, Wr_b, Ur_b, W_b, Aproj);
    recur_kernel<<<64, 256, 0, stream>>>(Aproj, uru_swz, U_b, attn, C_ws);
    nextmem_kernel<<<16, 256, 0, stream>>>(prevM, C_ws, questions, nm_swz, nm_b, next_mem);
}
